// Spiking_BasicBlock_13537736917752
// MI455X (gfx1250) — matrix operand order repaired and checked by reading, not compiled, not run on hardware
//
#include <hip/hip_runtime.h>

// Spiking BasicBlock for MI455X (gfx1250, wave32).
// - bit_decode collapsed through linear conv+BN: T=4 timestep convs -> 1 conv each.
// - Both convs as implicit-GEMM with v_wmma_f32_16x16x32_f16 (f32 accum).
// - Weight tiles staged into LDS by the Tensor Data Mover (tensor_load_to_lds),
//   input patches gathered with halo masking by the 8 waves in parallel.
// - BN fold + IF-neuron spike scan fused into the GEMM epilogue.

typedef _Float16 half_t;
typedef _Float16 half8  __attribute__((ext_vector_type(8)));
typedef _Float16 half16 __attribute__((ext_vector_type(16)));
typedef float    f32x8  __attribute__((ext_vector_type(8)));
typedef unsigned int u32x4 __attribute__((ext_vector_type(4)));
typedef int      i32x4 __attribute__((ext_vector_type(4)));
typedef int      i32x8 __attribute__((ext_vector_type(8)));

#define Bn 16
#define Tn 4
#define Cn 128
#define Hn 56
#define Wn 56
#define HW (Hn * Wn)          // 3136 pixels per image
#define KTOT (9 * Cn)         // 1152 = Cin * 9 taps, ordered K = tap*128 + ci
#define KC 32                 // K chunk per WMMA (f16: 16x16x32)
#define NCHUNK (KTOT / KC)    // 36
#define PXTILE 64             // output pixels per workgroup
#define NPXT (HW / PXTILE)    // 49 pixel tiles per image

// ---------------------------------------------------------------- prep kernels

__global__ void decode_kernel(const float* __restrict__ x,
                              float* __restrict__ xdec_f32,
                              half_t* __restrict__ xdec_f16) {
    size_t i = (size_t)blockIdx.x * blockDim.x + threadIdx.x;
    const size_t NTOT = (size_t)Bn * Cn * HW;
    if (i >= NTOT) return;
    size_t b   = i / ((size_t)Cn * HW);
    size_t rem = i % ((size_t)Cn * HW);
    const size_t stride = (size_t)Cn * HW;            // one timestep plane
    const float* px = x + (b * Tn) * stride + rem;    // x[b, t, c, h, w]
    float d = 8.0f * px[0] + 4.0f * px[stride] + 2.0f * px[2 * stride] + px[3 * stride];
    xdec_f32[i] = d;               // for identity path (exact)
    xdec_f16[i] = (half_t)d;       // GEMM B operand
}

// w[co, ci, kh, kw] (f32) -> wt[co][K = tap*128 + ci] (f16)
__global__ void wtrans_kernel(const float* __restrict__ w, half_t* __restrict__ wt) {
    int i = blockIdx.x * blockDim.x + threadIdx.x;
    if (i >= Cn * KTOT) return;
    int co  = i / KTOT;
    int rem = i % KTOT;
    int r   = rem / Cn;   // filter tap 0..8
    int ci  = rem % Cn;
    wt[i] = (half_t)w[(co * Cn + ci) * 9 + r];
}

// Fold BN: sb[0..127]=inv1/8, sb[128..]=beta1-mu1*inv1, sb[256..]=inv2/8, sb[384..]=beta2-mu2*inv2
__global__ void scalebias_kernel(const float* __restrict__ g1, const float* __restrict__ b1,
                                 const float* __restrict__ m1, const float* __restrict__ v1,
                                 const float* __restrict__ g2, const float* __restrict__ b2,
                                 const float* __restrict__ m2, const float* __restrict__ v2,
                                 float* __restrict__ sb) {
    int c = threadIdx.x;
    if (c < Cn) {
        float s1 = g1[c] * rsqrtf(v1[c] + 1e-5f);
        sb[c]          = s1 * 0.125f;
        sb[Cn + c]     = b1[c] - m1[c] * s1;
        float s2 = g2[c] * rsqrtf(v2[c] + 1e-5f);
        sb[2 * Cn + c] = s2 * 0.125f;
        sb[3 * Cn + c] = b2[c] - m2[c] * s2;
    }
}

// ------------------------------------------------- TDM weight-tile staging

// Issue TENSOR_LOAD_TO_LDS for a 2D tile: tile_dim0 = KC f16 (64B rows),
// tile_dim1 = Cn rows, global row stride = KTOT elements, data_size = 2B.
// LDS destination is filled linearly -> exactly ldsA[co*KC + k].
__device__ __forceinline__ void tdm_load_weights(const half_t* gsrc, unsigned lds_off) {
    unsigned long long ga = (unsigned long long)(uintptr_t)gsrc;
    // D# group 0: count=1 | lds_addr | global_addr[56:0] | type=2 ("image")
    u32x4 g0 = { 1u,
                 lds_off,
                 (unsigned)(ga & 0xFFFFFFFFu),
                 (unsigned)((ga >> 32) & 0x01FFFFFFu) | (2u << 30) };
    // D# group 1: wg_mask=0, data_size=1(2B); tensor_dim0=KTOT; tensor_dim1=Cn;
    //             tile_dim0=KC; tile_dim1=Cn; tensor_dim0_stride=KTOT.
    i32x8 g1 = { (int)0x00010000u,                    // [17:16] data_size = 1 (2 bytes)
                 (int)((KTOT & 0xFFFF) << 16),        // [63:48] tensor_dim0 lo16
                 (int)((Cn & 0xFFFF) << 16) | (KTOT >> 16),   // [95:80] dim1 lo16 | [79:64] dim0 hi16
                 (int)((KC & 0xFFFF) << 16) | (Cn >> 16),     // [127:112] tile_dim0 | [111:96] dim1 hi16
                 (int)(Cn & 0xFFFF),                  // [143:128] tile_dim1 ([159:144] tile_dim2=0)
                 (int)KTOT,                           // [191:160] tensor_dim0_stride lo32
                 0, 0 };                              // stride hi, dim1_stride (unused, 2D)
    i32x4 g2 = { 0, 0, 0, 0 };
    i32x4 g3 = { 0, 0, 0, 0 };
#if defined(__clang_major__) && (__clang_major__ >= 23)
    i32x8 g4 = { 0, 0, 0, 0, 0, 0, 0, 0 };
    __builtin_amdgcn_tensor_load_to_lds(g0, g1, g2, g3, g4, 0);
#else
    __builtin_amdgcn_tensor_load_to_lds(g0, g1, g2, g3, 0);
#endif
}

// ------------------------------------------------------- implicit-GEMM + WMMA

// PHASE 1: mem1 = conv1(xdec)*inv1/8 + b1c; spike scan -> out1dec (f16)
// PHASE 2: mem2 = conv2(out1dec)*inv2/8 + b2c + xdec/8; spike scan -> 4 output planes
template <int PHASE>
__global__ __launch_bounds__(256)
void conv_wmma_kernel(const half_t* __restrict__ Bsrc,     // [img][ci][hw] f16
                      const half_t* __restrict__ Wt,       // [co][K] f16
                      const float* __restrict__ scaleArr,  // [Cn]
                      const float* __restrict__ biasArr,   // [Cn]
                      const float* __restrict__ alpha,     // scalar
                      const float* __restrict__ identity,  // [img][c][hw] f32 (phase 2)
                      half_t* __restrict__ out1dec,        // phase 1 output
                      float* __restrict__ out)             // phase 2 output [b][t][c][hw]
{
    __shared__ __align__(64) half_t ldsA[Cn * KC];      // weights [co][k]   8 KB (TDM dest)
    __shared__ __align__(64) half_t ldsB[PXTILE * KC];  // patches [px][k]   4 KB

    const int tid    = threadIdx.x;
    const int wave   = tid >> 5;        // wave32
    const int lane   = tid & 31;
    const int img    = blockIdx.x / NPXT;
    const int pxBase = (blockIdx.x % NPXT) * PXTILE;

    const int waveCo = (wave >> 1) * 32;   // 4 co-groups
    const int wavePx = (wave & 1) * 32;    // 2 px-groups

    const unsigned ldsA_off = (unsigned)(uintptr_t)&ldsA[0];

    f32x8 acc[2][2];
#pragma unroll
    for (int i = 0; i < 2; ++i)
#pragma unroll
        for (int j = 0; j < 2; ++j) acc[i][j] = (f32x8)0.0f;

    // B-gather coordinates for this thread (one px column, 8 consecutive ci)
    const int bpx = tid & 63;     // 0..63
    const int kq  = tid >> 6;     // 0..3 -> ci sub-block of 8
    const int p   = pxBase + bpx;
    const int ph  = p / Wn;
    const int pw  = p % Wn;
    const half_t* imgBase = Bsrc + (size_t)img * Cn * HW;

    for (int kc = 0; kc < NCHUNK; ++kc) {
        const int kBase = kc * KC;
        const int r     = kBase / Cn;       // one filter tap per chunk
        const int ciB   = kBase % Cn;       // 0,32,64,96
        const int dh = r / 3 - 1, dw = r % 3 - 1;

        // ---- TDM: DMA this chunk's weight tile into ldsA (wave 0 issues)
        if (wave == 0)
            tdm_load_weights(Wt + kBase, ldsA_off);

        // ---- stage input patches -> ldsB[px][k] (zero-pad halo), all waves
        {
            const int hh = ph + dh, ww = pw + dw;
            const bool inb = (hh >= 0) & (hh < Hn) & (ww >= 0) & (ww < Wn);
            const int poff = hh * Wn + ww;
            __builtin_prefetch(imgBase + (size_t)(ciB + kq * 8) * HW + poff + 8 * HW, 0, 0);
            half8 v;
#pragma unroll
            for (int i = 0; i < 8; ++i) {
                const int ci = ciB + kq * 8 + i;
                v[i] = inb ? imgBase[(size_t)ci * HW + poff] : (half_t)0.0f;
            }
            *(half8*)&ldsB[bpx * KC + kq * 8] = v;
        }

        // wave 0 drains its tensor DMA, then the WG barrier releases everyone
        if (wave == 0)
            __builtin_amdgcn_s_wait_tensorcnt(0);
        __syncthreads();

        // ---- load fragments per CDNA5 wave32 WMMA layouts and multiply
        union H16 { half16 v; half8 h[2]; };
        const int m        = lane & 15;
        const int half_sel = lane >> 4;
        H16 afrag[2], bfrag[2];
#pragma unroll
        for (int t = 0; t < 2; ++t) {  // A: lanes0-15 k{0-7,16-23}; lanes16-31 k{8-15,24-31}
            const int row = waveCo + t * 16 + m;
            afrag[t].h[0] = *(const half8*)&ldsA[row * KC + half_sel * 8];
            afrag[t].h[1] = *(const half8*)&ldsA[row * KC + 16 + half_sel * 8];
        }
#pragma unroll
        for (int t = 0; t < 2; ++t) {  // B: lanes0-15 k{0-7,16-23}; lanes16-31 k{8-15,24-31}
            const int row = wavePx + t * 16 + m;
            bfrag[t].h[0] = *(const half8*)&ldsB[row * KC + half_sel * 8];
            bfrag[t].h[1] = *(const half8*)&ldsB[row * KC + 16 + half_sel * 8];
        }
#pragma unroll
        for (int i = 0; i < 2; ++i)
#pragma unroll
            for (int j = 0; j < 2; ++j)
                acc[i][j] = __builtin_amdgcn_wmma_f32_16x16x32_f16(
                    false, afrag[i].v, false, bfrag[j].v,
                    (short)0, acc[i][j], false, false);
        __syncthreads();   // protect LDS before next chunk's TDM/gather overwrite
    }

    // ---- epilogue: BN fold + IF-neuron spike scan (fully element-wise)
    const float th = alpha[0] * (8.0f / 15.0f);   // alpha * 2^(T-1)/(2^T-1)
    const int n  = lane & 15;
    const int mh = (lane >> 4) * 8;               // D layout: vgpr r -> M=r / M=r+8
#pragma unroll
    for (int i = 0; i < 2; ++i) {
#pragma unroll
        for (int j = 0; j < 2; ++j) {
#pragma unroll
            for (int rr = 0; rr < 8; ++rr) {
                const int co = waveCo + i * 16 + mh + rr;
                const int px = pxBase + wavePx + j * 16 + n;
                float mem = acc[i][j][rr] * scaleArr[co] + biasArr[co];
                if (PHASE == 2)
                    mem += identity[((size_t)img * Cn + co) * HW + px] * 0.125f;
                // spike_scan: factors 1 - 0.5^(T-t)
                const float fac[4] = {0.9375f, 0.875f, 0.75f, 0.5f};
                float sdec = 0.0f, wgt = 8.0f;
                float sp[4];
#pragma unroll
                for (int t = 0; t < 4; ++t) {
                    bool s = mem >= fac[t] * th;
                    sp[t] = s ? 1.0f : 0.0f;
                    if (s) mem -= th;
                    mem *= 2.0f;
                    sdec += s ? wgt : 0.0f;
                    wgt *= 0.5f;
                }
                if (PHASE == 1) {
                    out1dec[((size_t)img * Cn + co) * HW + px] = (half_t)(th * sdec);
                } else {
#pragma unroll
                    for (int t = 0; t < 4; ++t)
                        out[(((size_t)img * Tn + t) * Cn + co) * HW + px] = sp[t] * th;
                }
            }
        }
    }
}

// ------------------------------------------------------------------- launcher

extern "C" void kernel_launch(void* const* d_in, const int* in_sizes, int n_in,
                              void* d_out, int out_size, void* d_ws, size_t ws_size,
                              hipStream_t stream) {
    const float* x       = (const float*)d_in[0];
    const float* conv1_w = (const float*)d_in[1];
    const float* bn1_g   = (const float*)d_in[2];
    const float* bn1_b   = (const float*)d_in[3];
    const float* bn1_m   = (const float*)d_in[4];
    const float* bn1_v   = (const float*)d_in[5];
    const float* alpha1  = (const float*)d_in[6];
    const float* conv2_w = (const float*)d_in[7];
    const float* bn2_g   = (const float*)d_in[8];
    const float* bn2_b   = (const float*)d_in[9];
    const float* bn2_m   = (const float*)d_in[10];
    const float* bn2_v   = (const float*)d_in[11];
    const float* alpha2  = (const float*)d_in[12];
    float* out = (float*)d_out;

    // workspace carve-up (all offsets 256B-aligned): ~52 MB
    char* ws = (char*)d_ws;
    const size_t sz_xdec_f32 = (size_t)Bn * Cn * HW * sizeof(float);    // 25,690,112
    const size_t sz_xdec_f16 = (size_t)Bn * Cn * HW * sizeof(half_t);   // 12,845,056
    const size_t sz_wt       = (size_t)Cn * KTOT * sizeof(half_t);      //    294,912
    size_t off = 0;
    float*  xdec_f32 = (float*)(ws + off);  off += sz_xdec_f32;
    half_t* xdec_f16 = (half_t*)(ws + off); off += sz_xdec_f16;
    half_t* out1dec  = (half_t*)(ws + off); off += sz_xdec_f16;
    half_t* w1t      = (half_t*)(ws + off); off += sz_wt;
    half_t* w2t      = (half_t*)(ws + off); off += sz_wt;
    float*  sb       = (float*)(ws + off);  off += 4 * Cn * sizeof(float);

    const int nDec = Bn * Cn * HW;
    decode_kernel<<<(nDec + 255) / 256, 256, 0, stream>>>(x, xdec_f32, xdec_f16);

    const int nWt = Cn * KTOT;
    wtrans_kernel<<<(nWt + 255) / 256, 256, 0, stream>>>(conv1_w, w1t);
    wtrans_kernel<<<(nWt + 255) / 256, 256, 0, stream>>>(conv2_w, w2t);

    scalebias_kernel<<<1, 128, 0, stream>>>(bn1_g, bn1_b, bn1_m, bn1_v,
                                            bn2_g, bn2_b, bn2_m, bn2_v, sb);

    dim3 grid(Bn * NPXT);  // 784 workgroups, 256 threads (8 wave32) each
    conv_wmma_kernel<1><<<grid, 256, 0, stream>>>(
        xdec_f16, w1t, sb, sb + Cn, alpha1, nullptr, out1dec, nullptr);
    conv_wmma_kernel<2><<<grid, 256, 0, stream>>>(
        out1dec, w2t, sb + 2 * Cn, sb + 3 * Cn, alpha2, xdec_f32, nullptr, out);
}
